// LSTM_LabelSimilarity_54099408060900
// MI455X (gfx1250) — compile-verified, matrix operand order checked by reading, not run on hardware
//
#include <hip/hip_runtime.h>
#include <hip/hip_bf16.h>

// ---------------- problem constants ----------------
#define S_LEN 256
#define BATCH 64
#define EMB   300
#define EMB_P 320      // 300 padded to multiple of 32 for bf16 WMMA K
#define HID   512
#define G4H   2048     // 4*H
#define NCLS  46
#define NVAL  4
#define LROWS 184      // C*NV
#define LAB_P 256      // label rows padded to multiple of 128 (GEMM N tile)
#define NTOK  (S_LEN * BATCH)   // 16384
#define EPSV  1e-8f

typedef __attribute__((ext_vector_type(16))) __bf16 v16bf;
typedef __attribute__((ext_vector_type(8)))  float  v8f;

union FragBF { uint4 u[2]; v16bf v; };

__device__ __forceinline__ unsigned short f2bf(float f) {
  unsigned int u = __float_as_uint(f);
  u += 0x7FFFu + ((u >> 16) & 1u);          // round-to-nearest-even
  return (unsigned short)(u >> 16);
}
__device__ __forceinline__ float sigm(float x) { return 1.f / (1.f + expf(-x)); }
__device__ __forceinline__ float wredSum(float v) {
  for (int m = 16; m > 0; m >>= 1) v += __shfl_xor(v, m, 32);
  return v;
}
__device__ __forceinline__ float wredMax(float v) {
  for (int m = 16; m > 0; m >>= 1) v = fmaxf(v, __shfl_xor(v, m, 32));
  return v;
}

// ---------------------------------------------------------------------------
// Embedding gather: fp32 table -> bf16 [NTOK, EMB_P] (zero padded) + row norms
// ---------------------------------------------------------------------------
__global__ __launch_bounds__(256) void k_gather_embed(
    const int* __restrict__ seq, const float* __restrict__ emb,
    unsigned short* __restrict__ embB, float* __restrict__ en) {
  int tok = blockIdx.x * 8 + (threadIdx.x >> 5);
  if (tok >= NTOK) return;
  int lane = threadIdx.x & 31;
  int idx = seq[tok];
  const float* src = emb + (size_t)idx * EMB;
  unsigned short* dst = embB + (size_t)tok * EMB_P;
  float acc = 0.f;
  for (int e = lane; e < EMB_P; e += 32) {
    float v = (e < EMB) ? src[e] : 0.f;
    dst[e] = f2bf(v);
    acc += v * v;
  }
  acc = wredSum(acc);
  if (lane == 0) en[tok] = sqrtf(acc);
}

// Generic fp32 [rowsIn, Kin] -> bf16 [rowsOut, Kp] converter with zero padding
__global__ __launch_bounds__(256) void k_convert(
    const float* __restrict__ src, unsigned short* __restrict__ dst,
    int rowsIn, int rowsOut, int Kin, int Kp) {
  int idx = blockIdx.x * 256 + threadIdx.x;
  int total = rowsOut * Kp;
  if (idx >= total) return;
  int r = idx / Kp, k = idx - r * Kp;
  float v = (r < rowsIn && k < Kin) ? src[(size_t)r * Kin + k] : 0.f;
  dst[idx] = f2bf(v);
}

// Row L2 norms (one wave per row)
__global__ __launch_bounds__(256) void k_rownorm(
    const float* __restrict__ src, float* __restrict__ nrm, int rows, int K) {
  int r = blockIdx.x * 8 + (threadIdx.x >> 5);
  if (r >= rows) return;
  int lane = threadIdx.x & 31;
  float acc = 0.f;
  for (int e = lane; e < K; e += 32) {
    float v = src[(size_t)r * K + e];
    acc += v * v;
  }
  acc = wredSum(acc);
  if (lane == 0) nrm[r] = sqrtf(acc);
}

// Zero the recurrent state (h bf16 + c fp32)
__global__ __launch_bounds__(256) void k_zero_state(
    unsigned short* __restrict__ hB, float* __restrict__ c) {
  int idx = blockIdx.x * 256 + threadIdx.x;
  if (idx >= BATCH * HID) return;
  hB[idx] = 0;
  c[idx] = 0.f;
}

// bsum = b_ih + b_hh
__global__ __launch_bounds__(256) void k_bias_sum(
    const float* __restrict__ a, const float* __restrict__ b,
    float* __restrict__ o, int n) {
  int i = blockIdx.x * 256 + threadIdx.x;
  if (i < n) o[i] = a[i] + b[i];
}

// ---------------------------------------------------------------------------
// bf16 WMMA GEMM:  Out[M,N] = A[M,Kp] * Bt[N,Kp]^T (+ Cin[M,N])
// Block tile 64x128, 8 waves (2x4), each wave 32x32 = 2x2 WMMA 16x16x32 tiles.
// ---------------------------------------------------------------------------
#define LDT 40   // LDS row stride in bf16 elems (32 + 8 pad, keeps 16B align)
__global__ __launch_bounds__(256) void k_gemm_bf16_wmma(
    const unsigned short* __restrict__ A, const unsigned short* __restrict__ Bt,
    const float* __restrict__ Cin, float* __restrict__ Out,
    int M, int N, int Kp) {
  __shared__ unsigned short As[64 * LDT];
  __shared__ unsigned short Bs[128 * LDT];

  const int tid  = threadIdx.x;
  const int lane = tid & 31;
  const int wave = tid >> 5;
  const int wm = wave >> 2;           // 0..1 (M sub-tile of 32)
  const int wn = wave & 3;            // 0..3 (N sub-tile of 32)
  const int blockM = blockIdx.y * 64;
  const int blockN = blockIdx.x * 128;
  const int l15 = lane & 15;
  const int kg  = (lane >> 4) * 8;    // K-group select per ISA A-frag layout

  v8f acc[2][2];
  const v8f zero = {0.f, 0.f, 0.f, 0.f, 0.f, 0.f, 0.f, 0.f};
  acc[0][0] = zero; acc[0][1] = zero; acc[1][0] = zero; acc[1][1] = zero;

  for (int k0 = 0; k0 < Kp; k0 += 32) {
    {
      int r = tid >> 2, c8 = (tid & 3) * 8;
      uint4 v = *(const uint4*)(A + (size_t)(blockM + r) * Kp + k0 + c8);
      *(uint4*)(As + r * LDT + c8) = v;
    }
#pragma unroll
    for (int i = 0; i < 2; ++i) {
      int cid = tid + i * 256;
      int r = cid >> 2, c8 = (cid & 3) * 8;
      uint4 v = *(const uint4*)(Bt + (size_t)(blockN + r) * Kp + k0 + c8);
      *(uint4*)(Bs + r * LDT + c8) = v;
    }
    __syncthreads();

    FragBF fa[2], fb[2];
#pragma unroll
    for (int mi = 0; mi < 2; ++mi) {
      int row = wm * 32 + mi * 16 + l15;
      fa[mi].u[0] = *(const uint4*)(As + row * LDT + kg);
      fa[mi].u[1] = *(const uint4*)(As + row * LDT + kg + 16);
    }
#pragma unroll
    for (int ni = 0; ni < 2; ++ni) {
      int row = wn * 32 + ni * 16 + l15;
      fb[ni].u[0] = *(const uint4*)(Bs + row * LDT + kg);
      fb[ni].u[1] = *(const uint4*)(Bs + row * LDT + kg + 16);
    }
#pragma unroll
    for (int mi = 0; mi < 2; ++mi)
#pragma unroll
      for (int ni = 0; ni < 2; ++ni)
        acc[mi][ni] = __builtin_amdgcn_wmma_f32_16x16x32_bf16(
            false, fa[mi].v, false, fb[ni].v, (short)0, acc[mi][ni],
            false, false);
    __syncthreads();
  }

  const int rAdd = (lane >= 16) ? 8 : 0;
#pragma unroll
  for (int mi = 0; mi < 2; ++mi)
#pragma unroll
    for (int ni = 0; ni < 2; ++ni) {
      int colBase = blockN + wn * 32 + ni * 16 + l15;
      int rowBase = blockM + wm * 32 + mi * 16 + rAdd;
#pragma unroll
      for (int e = 0; e < 8; ++e) {
        size_t o = (size_t)(rowBase + e) * N + colBase;
        float v = acc[mi][ni][e];
        if (Cin) v += Cin[o];
        Out[o] = v;
      }
    }
}

// ---------------------------------------------------------------------------
// Fused LSTM step: one launch per timestep.
//   gates[b, g*512+j] = hIn @ Whh^T  (WMMA)  + Xt[b, g*512+j] + bsum[g*512+j]
// then the pointwise LSTM update, writing c (fp32) and hOut/hseq (bf16).
// Each block owns 128 j-columns across ALL FOUR gates so it can fuse the
// nonlinearity. A/B tiles are staged with global_load_async_to_lds_b128
// (ASYNCcnt) -- the CDNA5 async memory->LDS path, no VGPR bounce.
// Grid: HID/128 = 4 blocks, 256 threads (8 waves, 2x4).
// ---------------------------------------------------------------------------
__global__ __launch_bounds__(256) void k_lstm_step_fused(
    const unsigned short* __restrict__ hIn,  // [64, 512] bf16
    const unsigned short* __restrict__ Whh,  // [2048, 512] bf16 (gate-major)
    const float* __restrict__ Xt,            // [64, 2048] fp32 slab for step t
    const float* __restrict__ bsum,          // [2048] = b_ih + b_hh
    float* __restrict__ c,                   // [64, 512] fp32 cell state
    unsigned short* __restrict__ hOut,       // [64, 512] bf16
    unsigned short* __restrict__ hseq) {     // null or [64, 512] bf16 slab
  __shared__ unsigned short As[64 * LDT];        // h tile 64x32
  __shared__ unsigned short Bs[4 * 128 * LDT];   // 4 gate weight tiles 128x32

  const int tid  = threadIdx.x;
  const int lane = tid & 31;
  const int wave = tid >> 5;
  const int wm = wave >> 2;           // 0..1
  const int wn = wave & 3;            // 0..3
  const int j0 = blockIdx.x * 128;    // j-column base (0..511)
  const int l15 = lane & 15;
  const int kg  = (lane >> 4) * 8;

  v8f acc[4][2][2];
  const v8f zero = {0.f, 0.f, 0.f, 0.f, 0.f, 0.f, 0.f, 0.f};
#pragma unroll
  for (int g = 0; g < 4; ++g)
#pragma unroll
    for (int mi = 0; mi < 2; ++mi)
#pragma unroll
      for (int ni = 0; ni < 2; ++ni) acc[g][mi][ni] = zero;

  for (int k0 = 0; k0 < HID; k0 += 32) {
    // ---- async stage A (64x32): one 16B chunk per thread ----
    {
      int r = tid >> 2, c8 = (tid & 3) * 8;
      unsigned lo = (unsigned)(size_t)(As + r * LDT + c8);
      unsigned long long ga =
          (unsigned long long)(size_t)(hIn + (size_t)r * HID + k0 + c8);
      asm volatile("global_load_async_to_lds_b128 %0, %1, off"
                   :: "v"(lo), "v"(ga) : "memory");
    }
    // ---- async stage B: 4 gates x 128x32, 8 chunks per thread ----
#pragma unroll
    for (int g = 0; g < 4; ++g)
#pragma unroll
      for (int i = 0; i < 2; ++i) {
        int cid = tid + i * 256;
        int r = cid >> 2, c8 = (cid & 3) * 8;
        unsigned lo = (unsigned)(size_t)(Bs + (g * 128 + r) * LDT + c8);
        unsigned long long ga = (unsigned long long)(size_t)(
            Whh + (size_t)(g * HID + j0 + r) * HID + k0 + c8);
        asm volatile("global_load_async_to_lds_b128 %0, %1, off"
                     :: "v"(lo), "v"(ga) : "memory");
      }
    asm volatile("s_wait_asynccnt 0" ::: "memory");
    __syncthreads();

    FragBF fa[2];
#pragma unroll
    for (int mi = 0; mi < 2; ++mi) {
      int row = wm * 32 + mi * 16 + l15;
      fa[mi].u[0] = *(const uint4*)(As + row * LDT + kg);
      fa[mi].u[1] = *(const uint4*)(As + row * LDT + kg + 16);
    }
#pragma unroll
    for (int g = 0; g < 4; ++g) {
      FragBF fb[2];
#pragma unroll
      for (int ni = 0; ni < 2; ++ni) {
        int row = g * 128 + wn * 32 + ni * 16 + l15;
        fb[ni].u[0] = *(const uint4*)(Bs + row * LDT + kg);
        fb[ni].u[1] = *(const uint4*)(Bs + row * LDT + kg + 16);
      }
#pragma unroll
      for (int mi = 0; mi < 2; ++mi)
#pragma unroll
        for (int ni = 0; ni < 2; ++ni)
          acc[g][mi][ni] = __builtin_amdgcn_wmma_f32_16x16x32_bf16(
              false, fa[mi].v, false, fb[ni].v, (short)0, acc[g][mi][ni],
              false, false);
    }
    __syncthreads();
  }

  // ---- fused LSTM pointwise epilogue ----
  const int rAdd = (lane >= 16) ? 8 : 0;
#pragma unroll
  for (int mi = 0; mi < 2; ++mi)
#pragma unroll
    for (int ni = 0; ni < 2; ++ni) {
      int j = j0 + wn * 32 + ni * 16 + l15;       // 0..511
#pragma unroll
      for (int e = 0; e < 8; ++e) {
        int b = wm * 32 + mi * 16 + e + rAdd;     // 0..63
        const float* xr = Xt + (size_t)b * G4H;
        float gi = acc[0][mi][ni][e] + xr[j]             + bsum[j];
        float gf = acc[1][mi][ni][e] + xr[j + HID]       + bsum[j + HID];
        float gg = acc[2][mi][ni][e] + xr[j + 2 * HID]   + bsum[j + 2 * HID];
        float go = acc[3][mi][ni][e] + xr[j + 3 * HID]   + bsum[j + 3 * HID];
        int o = b * HID + j;
        float cn = sigm(gf) * c[o] + sigm(gi) * tanhf(gg);
        float h  = sigm(go) * tanhf(cn);
        c[o] = cn;
        unsigned short hb = f2bf(h);
        hOut[o] = hb;
        if (hseq) hseq[o] = hb;
      }
    }
}

// ---------------------------------------------------------------------------
// match[b,c] = max over s,v of num[s*B+b, c*4+v] / max(en*ln, eps)
// ---------------------------------------------------------------------------
__global__ __launch_bounds__(256) void k_cos_reduce(
    const float* __restrict__ num, const float* __restrict__ en,
    const float* __restrict__ ln, float* __restrict__ match) {
  int o = blockIdx.x * 8 + (threadIdx.x >> 5);
  if (o >= BATCH * NCLS) return;
  int lane = threadIdx.x & 31;
  int b = o / NCLS, c = o - b * NCLS;
  float m = -3.4e38f;
  for (int s = lane; s < S_LEN; s += 32) {
    int t = s * BATCH + b;
    float e = en[t];
    const float* np = num + (size_t)t * LAB_P + c * NVAL;
#pragma unroll
    for (int v = 0; v < NVAL; ++v) {
      float d = fmaxf(e * ln[c * NVAL + v], EPSV);
      m = fmaxf(m, np[v] / d);
    }
  }
  m = wredMax(m);
  if (lane == 0) match[o] = m;
}

// lstm_out[b,c] = out_b[c] + sum_k (feat[b,k]+lin_b[k]) * out_w[c,k]
__global__ __launch_bounds__(256) void k_head1(
    const float* __restrict__ feat, const float* __restrict__ lin_b,
    const float* __restrict__ out_w, const float* __restrict__ out_b,
    float* __restrict__ lout) {
  int idx = blockIdx.x * 256 + threadIdx.x;
  if (idx >= BATCH * NCLS) return;
  int b = idx / NCLS, c = idx - b * NCLS;
  float acc = out_b[c];
  const float* f = feat + (size_t)b * HID;
  const float* w = out_w + (size_t)c * HID;
  for (int k = 0; k < HID; ++k) acc += (f[k] + lin_b[k]) * w[k];
  lout[idx] = acc;
}

// out[b,c] = pred_b[c] + match[b,:] . pred_w[c,:46] + lout[b,:] . pred_w[c,46:]
__global__ __launch_bounds__(256) void k_head2(
    const float* __restrict__ match, const float* __restrict__ lout,
    const float* __restrict__ pred_w, const float* __restrict__ pred_b,
    float* __restrict__ out) {
  int idx = blockIdx.x * 256 + threadIdx.x;
  if (idx >= BATCH * NCLS) return;
  int b = idx / NCLS, c = idx - b * NCLS;
  const float* pw = pred_w + (size_t)c * (2 * NCLS);
  float acc = pred_b[c];
  for (int k = 0; k < NCLS; ++k) acc += match[b * NCLS + k] * pw[k];
  for (int k = 0; k < NCLS; ++k) acc += lout[b * NCLS + k] * pw[NCLS + k];
  out[idx] = acc;
}

// ---------------------------------------------------------------------------
extern "C" void kernel_launch(void* const* d_in, const int* in_sizes, int n_in,
                              void* d_out, int out_size, void* d_ws,
                              size_t ws_size, hipStream_t stream) {
  (void)in_sizes; (void)n_in; (void)out_size; (void)ws_size;

  const int*   seq      = (const int*)  d_in[0];
  const float* emb      = (const float*)d_in[1];
  const float* w_ih0    = (const float*)d_in[2];
  const float* w_hh0    = (const float*)d_in[3];
  const float* b_ih0    = (const float*)d_in[4];
  const float* b_hh0    = (const float*)d_in[5];
  const float* w_ih1    = (const float*)d_in[6];
  const float* w_hh1    = (const float*)d_in[7];
  const float* b_ih1    = (const float*)d_in[8];
  const float* b_hh1    = (const float*)d_in[9];
  const float* lin_w    = (const float*)d_in[10];
  const float* lin_b    = (const float*)d_in[11];
  const float* out_w    = (const float*)d_in[12];
  const float* out_b    = (const float*)d_in[13];
  const float* pred_w   = (const float*)d_in[14];
  const float* pred_b   = (const float*)d_in[15];
  const float* labelemb = (const float*)d_in[16];
  float* out = (float*)d_out;

  // ---- workspace carve-up ----
  char* ws = (char*)d_ws;
  size_t off = 0;
  auto alloc = [&](size_t bytes) -> char* {
    char* p = ws + off;
    off += (bytes + 255) & ~(size_t)255;
    return p;
  };
  unsigned short* embB  = (unsigned short*)alloc((size_t)NTOK * EMB_P * 2);
  float*          en    = (float*)         alloc((size_t)NTOK * 4);
  unsigned short* wih0B = (unsigned short*)alloc((size_t)G4H * EMB_P * 2);
  unsigned short* whh0B = (unsigned short*)alloc((size_t)G4H * HID * 2);
  unsigned short* wih1B = (unsigned short*)alloc((size_t)G4H * HID * 2);
  unsigned short* whh1B = (unsigned short*)alloc((size_t)G4H * HID * 2);
  unsigned short* linB  = (unsigned short*)alloc((size_t)HID * HID * 2);
  unsigned short* labB  = (unsigned short*)alloc((size_t)LAB_P * EMB_P * 2);
  float*          lnrm  = (float*)         alloc((size_t)LAB_P * 4);
  float*          X     = (float*)         alloc((size_t)NTOK * G4H * 4);
  float*          bsum0 = (float*)         alloc((size_t)G4H * 4);
  float*          bsum1 = (float*)         alloc((size_t)G4H * 4);
  float*          cbuf  = (float*)         alloc((size_t)BATCH * HID * 4);
  unsigned short* hPing = (unsigned short*)alloc((size_t)BATCH * HID * 2);
  unsigned short* hPong = (unsigned short*)alloc((size_t)BATCH * HID * 2);
  unsigned short* hseqB = (unsigned short*)alloc((size_t)NTOK * HID * 2);
  float*          num   = (float*)         alloc((size_t)NTOK * LAB_P * 4);
  float*          match = (float*)         alloc((size_t)BATCH * NCLS * 4);
  float*          feat  = (float*)         alloc((size_t)BATCH * HID * 4);
  float*          lout  = (float*)         alloc((size_t)BATCH * NCLS * 4);

  auto cdiv = [](int a, int b) { return (a + b - 1) / b; };

  // 1) embedding gather + norms; weight / label bf16 conversion; bias sums
  k_gather_embed<<<NTOK / 8, 256, 0, stream>>>(seq, emb, embB, en);
  k_convert<<<cdiv(G4H * EMB_P, 256), 256, 0, stream>>>(w_ih0, wih0B, G4H, G4H, EMB, EMB_P);
  k_convert<<<cdiv(G4H * HID,  256), 256, 0, stream>>>(w_hh0, whh0B, G4H, G4H, HID, HID);
  k_convert<<<cdiv(G4H * HID,  256), 256, 0, stream>>>(w_ih1, wih1B, G4H, G4H, HID, HID);
  k_convert<<<cdiv(G4H * HID,  256), 256, 0, stream>>>(w_hh1, whh1B, G4H, G4H, HID, HID);
  k_convert<<<cdiv(HID * HID,  256), 256, 0, stream>>>(lin_w, linB,  HID, HID, HID, HID);
  k_convert<<<cdiv(LAB_P * EMB_P, 256), 256, 0, stream>>>(labelemb, labB, LROWS, LAB_P, EMB, EMB_P);
  k_rownorm<<<cdiv(LROWS, 8), 256, 0, stream>>>(labelemb, lnrm, LROWS, EMB);
  k_bias_sum<<<cdiv(G4H, 256), 256, 0, stream>>>(b_ih0, b_hh0, bsum0, G4H);
  k_bias_sum<<<cdiv(G4H, 256), 256, 0, stream>>>(b_ih1, b_hh1, bsum1, G4H);

  // 2) layer0: hoisted input GEMM X = embed @ Wih0^T  [16384, 2048]
  k_gemm_bf16_wmma<<<dim3(G4H / 128, NTOK / 64), 256, 0, stream>>>(
      embB, wih0B, nullptr, X, NTOK, G4H, EMB_P);

  // 3) layer0 recurrence (single fused launch per step, ping-pong h)
  k_zero_state<<<cdiv(BATCH * HID, 256), 256, 0, stream>>>(hPing, cbuf);
  unsigned short* hCur = hPing;
  unsigned short* hNxt = hPong;
  for (int t = 0; t < S_LEN; ++t) {
    k_lstm_step_fused<<<dim3(HID / 128), 256, 0, stream>>>(
        hCur, whh0B, X + (size_t)t * BATCH * G4H, bsum0, cbuf, hNxt,
        hseqB + (size_t)t * BATCH * HID);
    unsigned short* tmp = hCur; hCur = hNxt; hNxt = tmp;
  }

  // 4) layer1: hoisted input GEMM over layer0 outputs
  k_gemm_bf16_wmma<<<dim3(G4H / 128, NTOK / 64), 256, 0, stream>>>(
      hseqB, wih1B, nullptr, X, NTOK, G4H, HID);

  // 5) layer1 recurrence (final hCur = h[-1] in bf16)
  k_zero_state<<<cdiv(BATCH * HID, 256), 256, 0, stream>>>(hPing, cbuf);
  hCur = hPing; hNxt = hPong;
  for (int t = 0; t < S_LEN; ++t) {
    k_lstm_step_fused<<<dim3(HID / 128), 256, 0, stream>>>(
        hCur, whh1B, X + (size_t)t * BATCH * G4H, bsum1, cbuf, hNxt, nullptr);
    unsigned short* tmp = hCur; hCur = hNxt; hNxt = tmp;
  }

  // 6) cosine-similarity numerator GEMM + max reduction
  k_gemm_bf16_wmma<<<dim3(LAB_P / 128, NTOK / 64), 256, 0, stream>>>(
      embB, labB, nullptr, num, NTOK, LAB_P, EMB_P);
  k_cos_reduce<<<cdiv(BATCH * NCLS, 8), 256, 0, stream>>>(num, en, lnrm, match);

  // 7) heads: feature = h[-1] @ lin_w^T (WMMA), then small scalar heads
  k_gemm_bf16_wmma<<<dim3(HID / 128, 1), 256, 0, stream>>>(
      hCur, linB, nullptr, feat, BATCH, HID, HID);
  k_head1<<<cdiv(BATCH * NCLS, 256), 256, 0, stream>>>(feat, lin_b, out_w, out_b, lout);
  k_head2<<<cdiv(BATCH * NCLS, 256), 256, 0, stream>>>(match, lout, pred_w, pred_b, out);
}
